// R_NEM_35424890257994
// MI455X (gfx1250) — matrix operand order repaired and checked by reading, not compiled, not run on hardware
//
#include <hip/hip_runtime.h>
#include <hip/hip_bf16.h>
#include <cstdint>

// ---------------------------------------------------------------------------
// Relational-NEM cell on MI455X (gfx1250, wave32, WMMA bf16 16x16x32).
// All GEMMs run on v_wmma_f32_16x16x32_bf16 with fused ReLU/tanh + LayerNorm
// epilogues inside the workgroup (each WG owns full output rows so LN's
// row-reduction stays in LDS). effect_sum is reduced in LDS (no atomics) by
// giving each WG two (b,i) groups of 7 j-rows.
// ---------------------------------------------------------------------------

#define BK     16384     // B*K rows
#define HV     250       // valid hidden width
#define HP     256       // padded hidden width
#define MX     576       // x feature width (already a multiple of 32)
#define HA_V   100       // attention hidden (valid)
#define HA_P   128       // attention hidden (padded)
#define KCORE  512       // padded 2*250 concat
#define KOUT   1088      // padded 250+250+576

typedef __attribute__((ext_vector_type(16))) __bf16 bfv16;
typedef __attribute__((ext_vector_type(8)))  float  fv8;

union FragU { bfv16 v; unsigned int u[8]; };

__device__ __forceinline__ unsigned short f2bf(float f) {
  union { float f; unsigned int u; } c; c.f = f;
  unsigned int u = c.u;
  u += 0x7FFFu + ((u >> 16) & 1u);      // round-to-nearest-even
  return (unsigned short)(u >> 16);
}

// A fragment (16x32 bf16, M x K): lanes 0-15 -> M=lane, K pairs {0,2,4,6,16,18,20,22};
// lanes 16-31 -> +8 on each K pair. Source is an LDS tile, row-major, even stride.
__device__ __forceinline__ bfv16 frag_A_lds(const unsigned short* As, int stride,
                                            int k0, int lane) {
  int m  = lane & 15;
  int kb = (lane & 16) ? 8 : 0;
  const unsigned short* row = As + m * stride + k0 + kb;
  FragU f;
#pragma unroll
  for (int p = 0; p < 8; ++p) {
    int koff = (p < 4) ? (2 * p) : (16 + 2 * (p - 4));
    f.u[p] = *(const unsigned int*)(row + koff);
  }
  return f.v;
}

// B fragment (32x16 bf16, K x N): lanes 0-15 -> N=lane, K pairs {0,2,4,6,16,18,20,22}; lanes 16-31 -> +8 on each K pair.
// Weights stored transposed [n][Kp] so each K pair is one aligned b32 load (L2 resident).
__device__ __forceinline__ bfv16 frag_B_glb(const unsigned short* __restrict__ Wt,
                                            int Kp, int n0, int k0, int lane) {
  int n  = lane & 15;
  int kb = (lane & 16) ? 8 : 0;
  const unsigned short* row = Wt + (size_t)(n0 + n) * Kp + k0 + kb;
  FragU f;
#pragma unroll
  for (int p = 0; p < 8; ++p) f.u[p] = *(const unsigned int*)(row + ((p < 4) ? (2 * p) : (16 + 2 * (p - 4))));
  return f.v;
}

#define WMMA_BF16(a, b, c) \
  __builtin_amdgcn_wmma_f32_16x16x32_bf16(false, (a), false, (b), (short)0, (c), false, false)

// LayerNorm stats for a 16-row LDS tile Cf[16][CP] over VC valid cols.
template <int CP, int NQ>
__device__ __forceinline__ void ln_stats(const float* Cf, int VC,
                                         float* redS, float* redQ,
                                         float* mu, float* rs, int tid) {
  int r = tid >> 4, t = tid & 15;
  float s = 0.f, q = 0.f;
#pragma unroll
  for (int i = 0; i < NQ; ++i) {
    int c = t + i * 16;
    if (c < VC) { float v = Cf[r * CP + c]; s += v; q += v * v; }
  }
  redS[r * 16 + t] = s; redQ[r * 16 + t] = q;
  __syncthreads();
  if (t == 0) {
    float a = 0.f, b = 0.f;
#pragma unroll
    for (int i = 0; i < 16; ++i) { a += redS[r * 16 + i]; b += redQ[r * 16 + i]; }
    float m = a / (float)VC;
    float v = b / (float)VC - m * m;
    mu[r] = m; rs[r] = rsqrtf(v + 1e-5f);
  }
  __syncthreads();
}

// ------------------------- conversion / packing kernels ---------------------

__global__ __launch_bounds__(256) void cvt_act_kernel(const float* __restrict__ src,
                                                      unsigned short* __restrict__ dst,
                                                      int C, int Cp, int total) {
  int i = blockIdx.x * 256 + threadIdx.x;
  if (i >= total) return;
  int r = i / Cp, c = i % Cp;
  float v = (c < C) ? src[(size_t)r * C + c] : 0.f;
  dst[i] = f2bf(v);
}

// Transpose-convert weight src[k][n] (row-major, srcN cols) to bf16 dst[n][Kp],
// with up to 3 K-segments placed at padded offsets {0,256,512}.
__global__ __launch_bounds__(256) void cvt_wt_kernel(const float* __restrict__ src,
                                                     unsigned short* __restrict__ dst,
                                                     int srcN, int Nvalid, int Kp,
                                                     int b0, int l0, int b1, int l1,
                                                     int b2, int l2, int total) {
  int i = blockIdx.x * 256 + threadIdx.x;
  if (i >= total) return;
  int n = i / Kp, k = i % Kp;
  float v = 0.f;
  if (n < Nvalid) {
    int base = -1, off = 0;
    if (k < 256)      { if (k < l0) { base = b0; off = k; } }
    else if (k < 512) { int kk = k - 256; if (kk < l1) { base = b1; off = kk; } }
    else              { int kk = k - 512; if (kk < l2) { base = b2; off = kk; } }
    if (base >= 0) v = src[(size_t)(base + off) * srcN + n];
  }
  dst[i] = f2bf(v);
}

// ------------------------------ enc layer -----------------------------------
// s1 = LN(relu(state @ enc_W + enc_b)); 16 rows per WG, full N=256 in LDS.
__global__ __launch_bounds__(256) void enc_kernel(
    const unsigned short* __restrict__ stateb, const unsigned short* __restrict__ encWt,
    const float* __restrict__ eb, const float* __restrict__ eg, const float* __restrict__ ebt,
    unsigned short* __restrict__ s1b) {
  __shared__ __align__(16) unsigned short As[16 * HP];
  __shared__ float Cf[16 * HP];
  __shared__ float redS[256], redQ[256], mu[16], rs[16];
  int tid = threadIdx.x;
  int row0 = blockIdx.x * 16;

  for (int it = tid; it < 16 * (HP / 8); it += 256) {           // 512 x uint4
    int r = it >> 5, cc = it & 31;
    *(uint4*)(&As[r * HP + cc * 8]) =
        *(const uint4*)(stateb + (size_t)(row0 + r) * HP + cc * 8);
  }
  __syncthreads();

  int wave = tid >> 5, lane = tid & 31;
  int cn = lane & 15, rb = (lane & 16) ? 8 : 0;
  int n0 = wave * 32, n1 = n0 + 16;
  fv8 acc0 = {}, acc1 = {};
#pragma unroll
  for (int kt = 0; kt < 8; ++kt) {
    bfv16 a  = frag_A_lds(As, HP, kt * 32, lane);
    bfv16 b0 = frag_B_glb(encWt, HP, n0, kt * 32, lane);
    bfv16 b1 = frag_B_glb(encWt, HP, n1, kt * 32, lane);
    acc0 = WMMA_BF16(a, b0, acc0);
    acc1 = WMMA_BF16(a, b1, acc1);
  }
  int c0 = n0 + cn, c1 = n1 + cn;
#pragma unroll
  for (int v = 0; v < 8; ++v) {
    Cf[(rb + v) * HP + c0] = (c0 < HV) ? fmaxf(acc0[v] + eb[c0], 0.f) : 0.f;
    Cf[(rb + v) * HP + c1] = (c1 < HV) ? fmaxf(acc1[v] + eb[c1], 0.f) : 0.f;
  }
  __syncthreads();
  ln_stats<HP, 16>(Cf, HV, redS, redQ, mu, rs, tid);
  int r = tid >> 4, t = tid & 15;
#pragma unroll
  for (int i = 0; i < 16; ++i) {
    int c = t + i * 16;
    float y = 0.f;
    if (c < HV) y = (Cf[r * HP + c] - mu[r]) * rs[r] * eg[c] + ebt[c];
    s1b[(size_t)(row0 + r) * HP + c] = f2bf(y);
  }
}

// ------------------- fused core + ctx + att + effect_sum --------------------
// One WG handles TWO (b,i) groups (7 j-rows each + 1 pad row = 16 rows), so the
// attention-weighted sum over j stays in LDS.
__global__ __launch_bounds__(256) void core_kernel(
    const unsigned short* __restrict__ s1b,
    const unsigned short* __restrict__ coreWt, const float* __restrict__ cb,
    const float* __restrict__ cg, const float* __restrict__ cbt,
    const unsigned short* __restrict__ ctxWt, const float* __restrict__ xb,
    const float* __restrict__ xg, const float* __restrict__ xbt,
    const unsigned short* __restrict__ att1Wt, const float* __restrict__ ab1,
    const float* __restrict__ ag, const float* __restrict__ abt,
    const float* __restrict__ aW2, const float* __restrict__ ab2,
    unsigned short* __restrict__ effb) {
  __shared__ __align__(16) unsigned short As[16 * KCORE];   // concat [fs|cs], 16KB
  __shared__ __align__(16) unsigned short Ac[16 * HP];      // core_out bf16, 8KB
  __shared__ float Cf[16 * HP];                             // f32 tile, 16KB
  __shared__ float Ca[16 * HA_P];                           // att tile, 8KB
  __shared__ float redS[256], redQ[256], mu[16], rs[16], aw[16];
  int tid = threadIdx.x;
  int g0 = blockIdx.x * 2, g1 = g0 + 1;

  // Stage concat A: row r -> group (r<8 ? g0 : g1), jj = r&7, +1 when >= ii (7 valid + 1 pad).
  for (int it = tid; it < 16 * (KCORE / 8); it += 256) {    // 1024 x uint4
    int r = it >> 6, cc = it & 63;
    uint4 val = {0u, 0u, 0u, 0u};
    int rr = r & 7;
    if (rr < 7) {
      int grp = (r < 8) ? g0 : g1;
      int bb = grp >> 3, ii = grp & 7;
      int jj = (rr < ii) ? rr : rr + 1;
      int srow = (cc < 32) ? (bb * 8 + ii) : (bb * 8 + jj);
      int c = cc & 31;
      val = *(const uint4*)(s1b + (size_t)srow * HP + c * 8);
    }
    *(uint4*)(&As[r * KCORE + cc * 8]) = val;
  }
  __syncthreads();

  int wave = tid >> 5, lane = tid & 31;
  int cn = lane & 15, rb = (lane & 16) ? 8 : 0;
  int n0 = wave * 32, n1 = n0 + 16, na = wave * 16;

  // core GEMM: K=512, N=256 (2 n-tiles per wave)
  {
    fv8 acc0 = {}, acc1 = {};
#pragma unroll
    for (int kt = 0; kt < 16; ++kt) {
      bfv16 a  = frag_A_lds(As, KCORE, kt * 32, lane);
      bfv16 b0 = frag_B_glb(coreWt, KCORE, n0, kt * 32, lane);
      bfv16 b1 = frag_B_glb(coreWt, KCORE, n1, kt * 32, lane);
      acc0 = WMMA_BF16(a, b0, acc0);
      acc1 = WMMA_BF16(a, b1, acc1);
    }
    int c0 = n0 + cn, c1 = n1 + cn;
#pragma unroll
    for (int v = 0; v < 8; ++v) {
      Cf[(rb + v) * HP + c0] = (c0 < HV) ? fmaxf(acc0[v] + cb[c0], 0.f) : 0.f;
      Cf[(rb + v) * HP + c1] = (c1 < HV) ? fmaxf(acc1[v] + cb[c1], 0.f) : 0.f;
    }
  }
  __syncthreads();
  ln_stats<HP, 16>(Cf, HV, redS, redQ, mu, rs, tid);
  {
    int r = tid >> 4, t = tid & 15;
#pragma unroll
    for (int i = 0; i < 16; ++i) {
      int c = t + i * 16;
      float y = 0.f;
      if (c < HV) y = (Cf[r * HP + c] - mu[r]) * rs[r] * cg[c] + cbt[c];
      Ac[r * HP + c] = f2bf(y);                 // core_out (bf16, pads = 0)
    }
  }
  __syncthreads();

  // ctx GEMM (N=256) + att GEMM (N=128) sharing the A fragment; K=256.
  {
    fv8 ac0 = {}, ac1 = {}, aa = {};
#pragma unroll
    for (int kt = 0; kt < 8; ++kt) {
      bfv16 a  = frag_A_lds(Ac, HP, kt * 32, lane);
      bfv16 b0 = frag_B_glb(ctxWt,  HP, n0, kt * 32, lane);
      bfv16 b1 = frag_B_glb(ctxWt,  HP, n1, kt * 32, lane);
      bfv16 b2 = frag_B_glb(att1Wt, HP, na, kt * 32, lane);
      ac0 = WMMA_BF16(a, b0, ac0);
      ac1 = WMMA_BF16(a, b1, ac1);
      aa  = WMMA_BF16(a, b2, aa);
    }
    int c0 = n0 + cn, c1 = n1 + cn, ca = na + cn;
#pragma unroll
    for (int v = 0; v < 8; ++v) {
      Cf[(rb + v) * HP + c0]   = (c0 < HV)   ? fmaxf(ac0[v] + xb[c0], 0.f) : 0.f;
      Cf[(rb + v) * HP + c1]   = (c1 < HV)   ? fmaxf(ac1[v] + xb[c1], 0.f) : 0.f;
      Ca[(rb + v) * HA_P + ca] = (ca < HA_V) ? tanhf(aa[v] + ab1[ca]) : 0.f;
    }
  }
  __syncthreads();

  // ctx LayerNorm (in place)
  ln_stats<HP, 16>(Cf, HV, redS, redQ, mu, rs, tid);
  {
    int r = tid >> 4, t = tid & 15;
    float m = mu[r], rr2 = rs[r];
#pragma unroll
    for (int i = 0; i < 16; ++i) {
      int c = t + i * 16;
      float y = 0.f;
      if (c < HV) y = (Cf[r * HP + c] - m) * rr2 * xg[c] + xbt[c];
      Cf[r * HP + c] = y;
    }
  }
  __syncthreads();

  // att LayerNorm (in place)
  ln_stats<HA_P, 8>(Ca, HA_V, redS, redQ, mu, rs, tid);
  {
    int r = tid >> 4, t = tid & 15;
#pragma unroll
    for (int i = 0; i < 8; ++i) {
      int c = t + i * 16;
      float y = 0.f;
      if (c < HA_V) y = (Ca[r * HA_P + c] - mu[r]) * rs[r] * ag[c] + abt[c];
      Ca[r * HA_P + c] = y;
    }
  }
  __syncthreads();

  // att2: per-row 100-dot + sigmoid
  if (tid < 16) {
    float s = ab2[0];
    for (int c = 0; c < HA_V; ++c) s += Ca[tid * HA_P + c] * aW2[c];
    aw[tid] = 1.f / (1.f + expf(-s));
  }
  __syncthreads();

  // effect_sum over the 7 j-rows of each group, all in LDS.
  {
    int c = tid;                                   // 0..255 columns
    float e0 = 0.f, e1 = 0.f;
#pragma unroll
    for (int j = 0; j < 7; ++j) {
      e0 += Cf[j * HP + c] * aw[j];
      e1 += Cf[(8 + j) * HP + c] * aw[8 + j];
    }
    effb[(size_t)g0 * HP + c] = f2bf(e0);
    effb[(size_t)g1 * HP + c] = f2bf(e1);
  }
}

// ------------------------------- out layer ----------------------------------
// new_state = [s1 | effect_sum | x] @ out_W + out_b  (K=1088 padded)
__global__ __launch_bounds__(256) void out_kernel(
    const unsigned short* __restrict__ s1b, const unsigned short* __restrict__ effb,
    const unsigned short* __restrict__ xb16, const unsigned short* __restrict__ outWt,
    const float* __restrict__ ob, float* __restrict__ out) {
  __shared__ __align__(16) unsigned short As[16 * KOUT];    // 34KB
  int tid = threadIdx.x;
  int row0 = blockIdx.x * 16;

  for (int it = tid; it < 16 * (KOUT / 8); it += 256) {     // 2176 x uint4
    int r = it / 136, cc = it % 136;
    size_t gr = (size_t)(row0 + r);
    uint4 val;
    if (cc < 32)      val = *(const uint4*)(s1b  + gr * HP + cc * 8);
    else if (cc < 64) val = *(const uint4*)(effb + gr * HP + (cc - 32) * 8);
    else              val = *(const uint4*)(xb16 + gr * MX + (cc - 64) * 8);
    *(uint4*)(&As[r * KOUT + cc * 8]) = val;
  }
  __syncthreads();

  int wave = tid >> 5, lane = tid & 31;
  int cn = lane & 15, rb = (lane & 16) ? 8 : 0;
  int n0 = wave * 32, n1 = n0 + 16;
  fv8 acc0 = {}, acc1 = {};
  for (int kt = 0; kt < 34; ++kt) {
    bfv16 a  = frag_A_lds(As, KOUT, kt * 32, lane);
    bfv16 b0 = frag_B_glb(outWt, KOUT, n0, kt * 32, lane);
    bfv16 b1 = frag_B_glb(outWt, KOUT, n1, kt * 32, lane);
    acc0 = WMMA_BF16(a, b0, acc0);
    acc1 = WMMA_BF16(a, b1, acc1);
  }
  int c0 = n0 + cn, c1 = n1 + cn;
#pragma unroll
  for (int v = 0; v < 8; ++v) {
    size_t gr = (size_t)(row0 + rb + v);
    if (c0 < HV) out[gr * HV + c0] = acc0[v] + ob[c0];
    if (c1 < HV) out[gr * HV + c1] = acc1[v] + ob[c1];
  }
}

// ------------------------------- host side ----------------------------------

extern "C" void kernel_launch(void* const* d_in, const int* in_sizes, int n_in,
                              void* d_out, int out_size, void* d_ws, size_t ws_size,
                              hipStream_t stream) {
  (void)in_sizes; (void)n_in; (void)out_size; (void)ws_size;
  const float* x      = (const float*)d_in[0];
  const float* state  = (const float*)d_in[1];
  const float* enc_W  = (const float*)d_in[2];
  const float* enc_b  = (const float*)d_in[3];
  const float* enc_g  = (const float*)d_in[4];
  const float* enc_bt = (const float*)d_in[5];
  const float* core_W = (const float*)d_in[6];
  const float* core_b = (const float*)d_in[7];
  const float* core_g = (const float*)d_in[8];
  const float* core_bt= (const float*)d_in[9];
  const float* ctx_W  = (const float*)d_in[10];
  const float* ctx_b  = (const float*)d_in[11];
  const float* ctx_g  = (const float*)d_in[12];
  const float* ctx_bt = (const float*)d_in[13];
  const float* att_W1 = (const float*)d_in[14];
  const float* att_b1 = (const float*)d_in[15];
  const float* att_g  = (const float*)d_in[16];
  const float* att_bt = (const float*)d_in[17];
  const float* att_W2 = (const float*)d_in[18];
  const float* att_b2 = (const float*)d_in[19];
  const float* out_W  = (const float*)d_in[20];
  const float* out_b  = (const float*)d_in[21];
  float* out = (float*)d_out;

  char* ws = (char*)d_ws;
  size_t off = 0;
  auto take = [&](size_t bytes) -> void* {
    void* p = ws + off;
    off += (bytes + 255) & ~(size_t)255;
    return p;
  };
  unsigned short* stateb = (unsigned short*)take((size_t)BK * HP * 2);
  unsigned short* xb     = (unsigned short*)take((size_t)BK * MX * 2);
  unsigned short* s1b    = (unsigned short*)take((size_t)BK * HP * 2);
  unsigned short* effb   = (unsigned short*)take((size_t)BK * HP * 2);
  unsigned short* encWt  = (unsigned short*)take((size_t)HP * HP * 2);
  unsigned short* coreWt = (unsigned short*)take((size_t)HP * KCORE * 2);
  unsigned short* ctxWt  = (unsigned short*)take((size_t)HP * HP * 2);
  unsigned short* att1Wt = (unsigned short*)take((size_t)HA_P * HP * 2);
  unsigned short* outWt  = (unsigned short*)take((size_t)HP * KOUT * 2);

  int tot;
  tot = BK * HP;
  cvt_act_kernel<<<(tot + 255) / 256, 256, 0, stream>>>(state, stateb, HV, HP, tot);
  tot = BK * MX;
  cvt_act_kernel<<<(tot + 255) / 256, 256, 0, stream>>>(x, xb, MX, MX, tot);
  tot = HP * HP;
  cvt_wt_kernel<<<(tot + 255) / 256, 256, 0, stream>>>(enc_W, encWt, HV, HV, HP,
                                                       0, HV, 0, 0, 0, 0, tot);
  tot = HP * KCORE;
  cvt_wt_kernel<<<(tot + 255) / 256, 256, 0, stream>>>(core_W, coreWt, HV, HV, KCORE,
                                                       0, HV, HV, HV, 0, 0, tot);
  tot = HP * HP;
  cvt_wt_kernel<<<(tot + 255) / 256, 256, 0, stream>>>(ctx_W, ctxWt, HV, HV, HP,
                                                       0, HV, 0, 0, 0, 0, tot);
  tot = HA_P * HP;
  cvt_wt_kernel<<<(tot + 255) / 256, 256, 0, stream>>>(att_W1, att1Wt, HA_V, HA_V, HP,
                                                       0, HV, 0, 0, 0, 0, tot);
  tot = HP * KOUT;
  cvt_wt_kernel<<<(tot + 255) / 256, 256, 0, stream>>>(out_W, outWt, HV, HV, KOUT,
                                                       0, HV, HV, HV, 2 * HV, MX, tot);

  enc_kernel<<<BK / 16, 256, 0, stream>>>(stateb, encWt, enc_b, enc_g, enc_bt, s1b);
  core_kernel<<<BK / 2, 256, 0, stream>>>(s1b,
                                          coreWt, core_b, core_g, core_bt,
                                          ctxWt, ctx_b, ctx_g, ctx_bt,
                                          att1Wt, att_b1, att_g, att_bt,
                                          att_W2, att_b2, effb);
  out_kernel<<<BK / 16, 256, 0, stream>>>(s1b, effb, xb, outWt, out_b, out);
}
